// HydModelNet_71399536328721
// MI455X (gfx1250) — matrix operand order repaired and checked by reading, not compiled, not run on hardware
//
#include <hip/hip_runtime.h>
#include <hip/hip_bf16.h>

typedef __attribute__((ext_vector_type(16))) _Float16 v16h;
typedef __attribute__((ext_vector_type(8)))  float    v8f;

namespace {
constexpr int kB = 512, kT = 365, kC = 8, kE = 32, kH = 256, kS = 16;
constexpr int kBL = 16 * 16 + 2 * 16;          // 288

// LDS row pitches (padded to break 512B-stride bank conflicts)
constexpr int LDX = 72;                        // x rows (64 used)
constexpr int LDH = 264;                       // h rows (256 used)
constexpr int LDB = 292;                       // b rows (288 used)

// packed-weight geometry (halfs); fragment blocks are 32 lanes x 16 halfs = 512
constexpr int W0_KT = 2,  W0_NT  = kH / 16;    // K padded 56->64
constexpr int W1_KT = 8,  W1_NT  = kH / 16;
constexpr int WIN_KT = 8, WIN_NT = 1;
constexpr int WOUT_KT = 8, WOUT_NT = kBL / 16; // 18
constexpr int SZ_W0   = W0_NT * W0_KT * 512;
constexpr int SZ_W1   = W1_NT * W1_KT * 512;
constexpr int SZ_WIN  = WIN_NT * WIN_KT * 512;
constexpr int SZ_WOUT = WOUT_NT * WOUT_KT * 512;
constexpr int OFF_W0 = 0;
constexpr int OFF_W1 = OFF_W0 + SZ_W0;
constexpr int OFF_W2 = OFF_W1 + SZ_W1;
constexpr int OFF_WIN = OFF_W2 + SZ_W1;
constexpr int OFF_WOUT = OFF_WIN + SZ_WIN;
} // namespace

// ---- pack fp32 weight [K][N] into f16 WMMA B-fragment order --------------
// lane n = l%16, halves j hold k = (l/16)*8 + j (j<8), 16 + (l/16)*8 + (j-8) (j>=8) within each 32x16 k-tile.
__global__ void pack_weights(const float* __restrict__ W, _Float16* __restrict__ dst,
                             int Kreal, int N, int ktiles, int total) {
  for (int idx = blockIdx.x * blockDim.x + threadIdx.x; idx < total;
       idx += gridDim.x * blockDim.x) {
    int blk = idx >> 9, rem = idx & 511;
    int lane = rem >> 4, j = rem & 15;
    int nt = blk / ktiles, kt = blk - nt * ktiles;
    int k = kt * 32 + ((j < 8) ? ((lane >> 4) * 8 + j) : (16 + (lane >> 4) * 8 + (j - 8)));
    int n = nt * 16 + (lane & 15);
    float v = (k < Kreal) ? W[(size_t)k * N + n] : 0.0f;
    dst[idx] = (_Float16)v;
  }
}

// ---- A-fragment from LDS row-major f16 activations (ISA 16-bit A layout) --
__device__ __forceinline__ v16h load_a_frag(const _Float16* sM, int ld, int m, int g, int k0) {
  const _Float16* row = sM + m * ld + k0;
  v16h a;
#pragma unroll
  for (int v = 0; v < 8; ++v) {
    int kb = (v < 4) ? (g * 8 + 2 * v)       // VGPR0-3: K = g*8 + 2v (+lo)
                     : (8 + g * 8 + 2 * v);  // VGPR4-7: K = 16 + g*8 + 2(v-4) (+lo)
    a[2 * v]     = row[kb];
    a[2 * v + 1] = row[kb + 1];
  }
  return a;
}

__device__ __forceinline__ v8f wmma_f16(v16h a, v16h b, v8f c) {
  return __builtin_amdgcn_wmma_f32_16x16x32_f16(false, a, false, b, (short)0, c,
                                                false, false);
}

// ---- persistent recurrent kernel: 1 block = 16 batch rows, all 365 steps --
// 512 threads = 16 wave32s; each wave owns one 16-wide N-tile of H=256.
__global__ __launch_bounds__(512) void hyd_persist(
    const float* __restrict__ clim, const float* __restrict__ rain,
    const float* __restrict__ enc,
    const float* __restrict__ b0, const float* __restrict__ b1,
    const float* __restrict__ b2, const float* __restrict__ b_in,
    const float* __restrict__ W_et, const float* __restrict__ b_et,
    const float* __restrict__ b_out,
    const float* __restrict__ W_init, const float* __restrict__ b_init,
    const _Float16* __restrict__ wpack, float* __restrict__ out) {
  __shared__ _Float16 sX[16 * LDX];
  __shared__ _Float16 sH[16 * LDH];
  __shared__ float sB[16 * LDB];
  __shared__ float sA[16 * 16];
  __shared__ float sSt[16 * 16];
  __shared__ float sEt[16];
  __shared__ float sClim[2][16][8];   // async double-buffered climate staging

  const int tid = threadIdx.x;
  const int lane = tid & 31;
  const int wave = tid >> 5;
  const int m16 = lane & 15;   // tile column (N) / row group select
  const int g = lane >> 4;
  const int rowBase = blockIdx.x * 16;
  const int rr = tid >> 4;     // elementwise: batch row 0..15 (tid < 256)
  const int ss = tid & 15;     // elementwise: store index 0..15

  const _Float16* w0p = wpack + OFF_W0;
  const _Float16* w1p = wpack + OFF_W1;
  const _Float16* w2p = wpack + OFF_W2;
  const _Float16* winp = wpack + OFF_WIN;
  const _Float16* woutp = wpack + OFF_WOUT;

  // per-lane async-staging addresses: lane -> (row, 16B chunk)
  const int cR = lane >> 1, cP = lane & 1;
  const unsigned ldsClim0 = (unsigned)(size_t)&sClim[0][cR][cP * 4];
  const unsigned ldsClim1 = (unsigned)(size_t)&sClim[1][cR][cP * 4];

  // stores0 = exp(enc @ W_init + b_init)   (tiny: one thread per element)
  if (tid < 256) {
    float acc = b_init[ss];
    const float* erow = enc + (size_t)(rowBase + rr) * kE;
#pragma unroll
    for (int e = 0; e < kE; ++e) acc = fmaf(erow[e], W_init[e * kS + ss], acc);
    sSt[rr * 16 + ss] = __expf(acc);
  }

  // kick off async climate load for t = 0 into buffer 0
  if (wave == 0) {
    unsigned gOff = (unsigned)((((size_t)(rowBase + cR) * kT + 0) * kC + cP * 4) * sizeof(float));
    asm volatile("global_load_async_to_lds_b128 %0, %1, %2"
                 :: "v"(ldsClim0), "v"(gOff), "s"(clim) : "memory");
  }
  __syncthreads();

  for (int t = 0; t < kT; ++t) {
    // issue async prefetch of next step's climate into the other buffer,
    // then wait so that *this* step's buffer (issued last iteration) is ready
    if (wave == 0) {
      int tn = (t + 1 < kT) ? (t + 1) : (kT - 1);
      unsigned gOff = (unsigned)((((size_t)(rowBase + cR) * kT + tn) * kC + cP * 4) * sizeof(float));
      asm volatile("global_load_async_to_lds_b128 %0, %1, %2"
                   :: "v"(((t + 1) & 1) ? ldsClim1 : ldsClim0), "v"(gOff), "s"(clim)
                   : "memory");
      asm volatile("s_wait_asynccnt 0x1" ::: "memory");
      // keep L2-resident weights warm in near caches (global_prefetch_b8)
      __builtin_prefetch(w1p + (size_t)lane * W1_KT * 512, 0, 1);
    }
    __syncthreads();

    // ---- x = [clim | enc | log(clamp(stores,0.1))] as f16, zero-pad to 64
    for (int idx = tid; idx < 16 * 64; idx += 512) {
      int r = idx >> 6, c = idx & 63;
      int grow = rowBase + r;
      float v;
      if (c < kC)                v = sClim[t & 1][r][c];
      else if (c < kC + kE)      v = enc[(size_t)grow * kE + (c - kC)];
      else if (c < kC + kE + kS) v = __logf(fmaxf(sSt[r * 16 + (c - kC - kE)], 0.1f));
      else                       v = 0.0f;
      sX[r * LDX + c] = (_Float16)v;
    }
    __syncthreads();

    // ---- layer 0: h = relu(x @ W0 + b0), [16,64]x[64,256], 1 tile/wave ----
    {
      const int nt = wave;
      v8f acc = {};
#pragma unroll
      for (int kt = 0; kt < W0_KT; ++kt) {
        v16h a = load_a_frag(sX, LDX, m16, g, kt * 32);
        v16h bf = *(const v16h*)(w0p + ((size_t)(nt * W0_KT + kt) * 32 + lane) * 16);
        acc = wmma_f16(a, bf, acc);
      }
      float bv = b0[nt * 16 + m16];
#pragma unroll
      for (int r = 0; r < 8; ++r)
        sH[(r + 8 * g) * LDH + nt * 16 + m16] = (_Float16)fmaxf(acc[r] + bv, 0.f);
    }
    __syncthreads();

    // ---- layers 1 & 2: h = relu(h @ W + b), 8 k-steps, in-place in sH ----
    for (int L = 0; L < 2; ++L) {
      const _Float16* wp = (L == 0) ? w1p : w2p;
      const float* bias = (L == 0) ? b1 : b2;
      const int nt = wave;
      v8f acc = {};
#pragma unroll
      for (int kt = 0; kt < W1_KT; ++kt) {
        v16h a = load_a_frag(sH, LDH, m16, g, kt * 32);
        v16h bf = *(const v16h*)(wp + ((size_t)(nt * W1_KT + kt) * 32 + lane) * 16);
        acc = wmma_f16(a, bf, acc);
      }
      __syncthreads();   // all reads of h done before in-place overwrite
      float bv = bias[nt * 16 + m16];
#pragma unroll
      for (int r = 0; r < 8; ++r)
        sH[(r + 8 * g) * LDH + nt * 16 + m16] = (_Float16)fmaxf(acc[r] + bv, 0.f);
      __syncthreads();
    }

    // ---- heads: W_out (18 tiles / 16 waves), W_in (wave 15), ET (VALU) ----
    {
      int ntO[2] = {wave, wave + 16};
      const int nO = (wave < 2) ? 2 : 1;
      v8f accO[2] = {{}, {}};
      v8f accI = {};
#pragma unroll
      for (int kt = 0; kt < WOUT_KT; ++kt) {
        v16h a = load_a_frag(sH, LDH, m16, g, kt * 32);
        for (int i = 0; i < nO; ++i) {
          v16h bf = *(const v16h*)(woutp + ((size_t)(ntO[i] * WOUT_KT + kt) * 32 + lane) * 16);
          accO[i] = wmma_f16(a, bf, accO[i]);
        }
        if (wave == 15) {
          v16h bf = *(const v16h*)(winp + ((size_t)kt * 32 + lane) * 16);
          accI = wmma_f16(a, bf, accI);
        }
      }
      // ET: 256-length dot per row, 16 lanes x 16 elems + width-16 butterfly
      if (tid < 256) {
        float ep = 0.f;
#pragma unroll
        for (int i = 0; i < 16; ++i)
          ep = fmaf((float)sH[rr * LDH + ss * 16 + i], W_et[ss * 16 + i], ep);
#pragma unroll
        for (int msk = 1; msk < 16; msk <<= 1) ep += __shfl_xor(ep, msk, 16);
        if (ss == 0) {
          float x = ep + b_et[0];
          sEt[rr] = fmaxf(x, 0.f) + log1pf(__expf(-fabsf(x)));  // softplus
        }
      }
      // sigmoid gate epilogue (fp32 in LDS)
      for (int i = 0; i < nO; ++i) {
        float bv = b_out[ntO[i] * 16 + m16];
#pragma unroll
        for (int r = 0; r < 8; ++r) {
          float x = accO[i][r] + bv;
          sB[(r + 8 * g) * LDB + ntO[i] * 16 + m16] = 1.0f / (1.0f + __expf(-x));
        }
      }
      if (wave == 15) {
        float bv = b_in[m16];
#pragma unroll
        for (int r = 0; r < 8; ++r) sA[(r + 8 * g) * 16 + m16] = accI[r] + bv;
      }
    }
    __syncthreads();

    // ---- store dynamics: one thread per (row, store), shuffle reductions --
    if (tid < 256) {
      float lg = sA[rr * 16 + ss];
      float mx = lg;
#pragma unroll
      for (int msk = 1; msk < 16; msk <<= 1) mx = fmaxf(mx, __shfl_xor(mx, msk, 16));
      float ex = __expf(lg - mx);
      float sm = ex;
#pragma unroll
      for (int msk = 1; msk < 16; msk <<= 1) sm += __shfl_xor(sm, msk, 16);
      float afrac = ex / sm;

      float cr = fmaxf(rain[(size_t)(rowBase + rr) * kT + t] - sEt[rr], 0.f);
      float st = sSt[rr * 16 + ss] + afrac * cr;

#pragma unroll
      for (int d = 0; d < kS; ++d) {        // sequential inter-store flow
        float fb = sB[rr * LDB + d * 16 + ss] * st;
        float fs = fb;
#pragma unroll
        for (int msk = 1; msk < 16; msk <<= 1) fs += __shfl_xor(fs, msk, 16);
        st = st - fb + ((ss == d) ? fs : 0.f);
      }
      st *= (1.f - sB[rr * LDB + 256 + ss]);            // escape flux
      float flow = sB[rr * LDB + 272 + ss] * st;        // outflow
      st -= flow;
      float fsum = flow;
#pragma unroll
      for (int msk = 1; msk < 16; msk <<= 1) fsum += __shfl_xor(fsum, msk, 16);
      if (ss == 0) out[(size_t)t * kB + rowBase + rr] = fsum;
      sSt[rr * 16 + ss] = st;
    }
    __syncthreads();
  }
}

extern "C" void kernel_launch(void* const* d_in, const int* in_sizes, int n_in,
                              void* d_out, int out_size, void* d_ws, size_t ws_size,
                              hipStream_t stream) {
  const float* clim   = (const float*)d_in[0];
  const float* rain   = (const float*)d_in[1];
  const float* enc    = (const float*)d_in[2];
  const float* W0     = (const float*)d_in[3];
  const float* b0     = (const float*)d_in[4];
  const float* W1     = (const float*)d_in[5];
  const float* b1     = (const float*)d_in[6];
  const float* W2     = (const float*)d_in[7];
  const float* b2     = (const float*)d_in[8];
  const float* W_in   = (const float*)d_in[9];
  const float* b_in   = (const float*)d_in[10];
  const float* W_et   = (const float*)d_in[11];
  const float* b_et   = (const float*)d_in[12];
  const float* W_out  = (const float*)d_in[13];
  const float* b_out  = (const float*)d_in[14];
  const float* W_init = (const float*)d_in[15];
  const float* b_init = (const float*)d_in[16];
  float* out = (float*)d_out;
  _Float16* wpack = (_Float16*)d_ws;    // ~440 KB of f16 packed weights

  pack_weights<<<64, 256, 0, stream>>>(W0,    wpack + OFF_W0,   kC + kE + kS, kH,  W0_KT,   SZ_W0);
  pack_weights<<<64, 256, 0, stream>>>(W1,    wpack + OFF_W1,   kH,           kH,  W1_KT,   SZ_W1);
  pack_weights<<<64, 256, 0, stream>>>(W2,    wpack + OFF_W2,   kH,           kH,  W1_KT,   SZ_W1);
  pack_weights<<<64, 256, 0, stream>>>(W_in,  wpack + OFF_WIN,  kH,           kS,  WIN_KT,  SZ_WIN);
  pack_weights<<<64, 256, 0, stream>>>(W_out, wpack + OFF_WOUT, kH,           kBL, WOUT_KT, SZ_WOUT);

  hyd_persist<<<kB / 16, 512, 0, stream>>>(clim, rain, enc, b0, b1, b2, b_in,
                                           W_et, b_et, b_out, W_init, b_init,
                                           wpack, out);
}
